// BlockCirculantProcess_60129542454
// MI455X (gfx1250) — matrix operand order repaired and checked by reading, not compiled, not run on hardware
//
#include <hip/hip_runtime.h>
#include <math.h>

typedef __attribute__((ext_vector_type(16))) _Float16 v16h;
typedef __attribute__((ext_vector_type(8)))  float    v8f;
typedef __attribute__((ext_vector_type(8)))  unsigned int v8u;

// Problem constants (from reference)
//   BATCH=4096, IN=OUT=4096, B_SIZE=128, K_HALF=65, K_TRUNC=48, K_IN=K_OUT=32
// Factorization:
//   X  = x_blocks(131072 x 128) @ Fb(128 x 96)            [96 = 48 Re bins | 48 Im bins]
//   Y_k = Xcat_k(4096 x 64) @ U_k(64 x 64)  for k=0..47   [64 = (Re|Im) x 32 blocks]
//   out = Ycat(131072 x 96) @ G(96 x 128)
//
// ws layout (dword offsets):
#define FB_OFF    0                      // Fb fragments:  4 ktiles * 6 ntiles * 256 dw = 6144
#define U_OFF     6144                   // U fragments :  48 bins * (2 kt * 4 nt) * 256 dw = 98304
#define G_OFF     104448                 // G fragments :  3 ktiles * 8 ntiles * 256 dw = 6144
#define X2_OFF_DW 110592                 // X2: 48*4096*64 halves = 6291456 dw
#define Y_OFF_DW  (110592 + 6291456)     // Y : 131072*96 halves  = 6291456 dw
// total ws needed = (110592 + 2*6291456)*4 bytes ~= 50.8 MB

__device__ inline v8f wmma_f16(v16h a, v16h b, v8f c) {
  return __builtin_amdgcn_wmma_f32_16x16x32_f16(
      /*neg_a=*/false, a, /*neg_b=*/false, b,
      /*c_mod=*/(short)0, c, /*reuse_a=*/false, /*reuse_b=*/false);
}

// A fragment (16x32 f16, row-major source, row stride ld elements)
// Layout: lanes 0-15 -> M=lane, K offsets {0..15}; lanes 16-31 -> M=lane-16, K {16..31}
__device__ inline v16h load_A_f16(const _Float16* __restrict__ src, int ld, int lane) {
  int m  = lane & 15;
  int kb = (lane < 16) ? 0 : 16;
  const _Float16* p = src + m * ld + kb;
  v16h a;
#pragma unroll
  for (int v = 0; v < 8; ++v) {
    int k = 2 * v;
    a[2 * v]     = p[k];
    a[2 * v + 1] = p[k + 1];
  }
  return a;
}

// Same A fragment but converting from f32 source on the fly
__device__ inline v16h load_A_f32(const float* __restrict__ src, int ld, int lane) {
  int m  = lane & 15;
  int kb = (lane < 16) ? 0 : 16;
  const float* p = src + m * ld + kb;
  v16h a;
#pragma unroll
  for (int v = 0; v < 8; ++v) {
    int k = 2 * v;
    a[2 * v]     = (_Float16)p[k];
    a[2 * v + 1] = (_Float16)p[k + 1];
  }
  return a;
}

// B fragment (pre-packed layout: per tile 32 lanes * 8 dwords, lane-major).
// Works for both global (ws) and LDS sources; addrspace inferred after inlining.
__device__ inline v16h load_B_frag(const unsigned int* base, int lane) {
  v8u t = *(const v8u*)(base + lane * 8);
  v16h b;
  __builtin_memcpy(&b, &t, sizeof(b));
  return b;
}

__device__ inline unsigned int pack2(float a, float b) {
  union { _Float16 h[2]; unsigned int u; } p;
  p.h[0] = (_Float16)a;
  p.h[1] = (_Float16)b;
  return p.u;
}

// ---------------------------------------------------------------------------
// Kernel 0: build Fb / U_k / G directly in WMMA B-fragment layout.
// One thread per packed dword (f16 pair). Total dwords = 110592 = 432*256.
// B-fragment element map: n = nt*16 + (lane&15); K = kt*32 + (lane<16?0:16) + 2v + h
// ---------------------------------------------------------------------------
__global__ void bc_build_mats(const float* __restrict__ Wr,
                              const float* __restrict__ Wi,
                              unsigned int* __restrict__ ws) {
  const int tid = blockIdx.x * 256 + threadIdx.x;
  const float W0 = 6.28318530717958647692f / 128.0f;  // 2*pi/B
  float vals[2];
  unsigned int* dst;

  if (tid < 6144) {
    // Fb: K=t (time 0..127), N=c (0..95): c<48 -> cos(w*c*t); else -> -sin(w*(c-48)*t)
    int d = tid, tile = d >> 8, w = d & 255, lane = w >> 3, v = w & 7;
    int kt = tile / 6, nt = tile % 6;
    int n = nt * 16 + (lane & 15);
    int Kb = kt * 32 + ((lane < 16) ? 0 : 16) + 2 * v;
#pragma unroll
    for (int h = 0; h < 2; ++h) {
      int t = Kb + h;
      vals[h] = (n < 48) ? cosf(W0 * (float)(n * t))
                         : -sinf(W0 * (float)((n - 48) * t));
    }
    dst = ws + FB_OFF + tile * 256 + lane * 8 + v;
  } else if (tid < 6144 + 98304) {
    // U_k: rows a=(Re|Im,j), cols c=(YrI|YiI,i).  U = [[Cr,-Ci],[Ci,Cr]]
    int d = tid - 6144;
    int bin = d >> 11, rem = d & 2047;
    int tile = rem >> 8, w = rem & 255, lane = w >> 3, v = w & 7;
    int nt = tile & 3, kt = tile >> 2;
    int n = nt * 16 + (lane & 15);
    int i = n & 31;
    bool creal = (n < 32);
    int Kb = kt * 32 + ((lane < 16) ? 0 : 16) + 2 * v;
#pragma unroll
    for (int h = 0; h < 2; ++h) {
      int a = Kb + h;
      int j = a & 31;
      bool areal = (a < 32);
      float cr = Wr[(i * 32 + j) * 65 + bin];
      float ci = Wi[(i * 32 + j) * 65 + bin];
      vals[h] = creal ? (areal ? cr : ci) : (areal ? -ci : cr);
    }
    dst = ws + U_OFF + bin * 2048 + tile * 256 + lane * 8 + v;
  } else if (tid < 110592) {
    // G: rows p=(part,k) 0..95, cols t 0..127:
    //   part0: wk*cos(w*k*t); part1: -wk*sin(w*k*t); wk = (k==0?1:2)/128
    int d = tid - 104448;
    int tile = d >> 8, w = d & 255, lane = w >> 3, v = w & 7;
    int kt = tile / 8, nt = tile % 8;
    int t = nt * 16 + (lane & 15);
    int Kb = kt * 32 + ((lane < 16) ? 0 : 16) + 2 * v;
#pragma unroll
    for (int h = 0; h < 2; ++h) {
      int p = Kb + h;
      int part = p / 48, k = p % 48;
      float wk = ((k == 0) ? 1.0f : 2.0f) * (1.0f / 128.0f);
      vals[h] = (part == 0) ? wk * cosf(W0 * (float)(k * t))
                            : -wk * sinf(W0 * (float)(k * t));
    }
    dst = ws + G_OFF + tile * 256 + lane * 8 + v;
  } else {
    return;
  }

  union { _Float16 h[2]; unsigned int u; } pk;
  pk.h[0] = (_Float16)vals[0];
  pk.h[1] = (_Float16)vals[1];
  *dst = pk.u;
}

// ---------------------------------------------------------------------------
// Kernel 1: forward DFT GEMM.  X(131072 x 96) = x(131072 x 128) @ Fb(128 x 96)
// Fb fragments staged in LDS (24 KB) once per block; each wave owns 2 M-tiles
// so every ds_load_b128 B-fragment feeds two independent WMMAs.
// __launch_bounds__(128,1): full VGPR budget, no accumulator spills.
// Output written bin-major with packed b128 stores: X2[k][b][part*32 + j] (f16)
// ---------------------------------------------------------------------------
__global__ void __launch_bounds__(128, 1)
bc_fwd_dft(const float* __restrict__ x, unsigned int* __restrict__ ws) {
  __shared__ unsigned int sB[6144];
  {
    const uint4* src = (const uint4*)(ws + FB_OFF);
    uint4* dstv = (uint4*)sB;
    for (int i = threadIdx.x; i < 1536; i += 128) dstv[i] = src[i];
  }
  __syncthreads();

  const int lane = threadIdx.x & 31;
  const int wave = threadIdx.x >> 5;
  const int mt0  = (blockIdx.x * 4 + wave) * 2;  // 2 M-tiles per wave
  _Float16* X2 = (_Float16*)(ws + X2_OFF_DW);

  __builtin_prefetch(x + (size_t)mt0 * 2048 + 2048, 0, 1);

  v8f acc[2][6] = {};
#pragma unroll
  for (int kt = 0; kt < 4; ++kt) {
    v16h a0 = load_A_f32(x + (mt0 + 0) * 2048 + kt * 32, 128, lane);
    v16h a1 = load_A_f32(x + (mt0 + 1) * 2048 + kt * 32, 128, lane);
#pragma unroll
    for (int nt = 0; nt < 6; ++nt) {
      v16h b = load_B_frag(sB + (kt * 6 + nt) * 256, lane);
      acc[0][nt] = wmma_f16(a0, b, acc[0][nt]);
      acc[1][nt] = wmma_f16(a1, b, acc[1][nt]);
    }
  }

  const int nc  = lane & 15;
  const int mhi = (lane < 16) ? 0 : 8;
#pragma unroll
  for (int u = 0; u < 2; ++u) {
    int r0 = (mt0 + u) * 16 + mhi;       // 8 consecutive rows, same b / part run
    int bb = r0 >> 5, j0 = r0 & 31;      // j0 in {0,8,16,24} -> 16B aligned
#pragma unroll
    for (int nt = 0; nt < 6; ++nt) {
      int c = nt * 16 + nc;
      int k = c % 48, part = c / 48;
      uint4 pkt;
      pkt.x = pack2(acc[u][nt][0], acc[u][nt][1]);
      pkt.y = pack2(acc[u][nt][2], acc[u][nt][3]);
      pkt.z = pack2(acc[u][nt][4], acc[u][nt][5]);
      pkt.w = pack2(acc[u][nt][6], acc[u][nt][7]);
      *(uint4*)(X2 + k * 262144 + bb * 64 + part * 32 + j0) = pkt;
    }
  }
}

// ---------------------------------------------------------------------------
// Kernel 2: per-bin complex mixing. Block owns one bin (U_k staged in 8 KB LDS,
// shared by all 4 waves); each wave owns 2 batch-tiles:
//   Y_k(16 x 64) += Xcat_k(16 x 64) @ U_k(64 x 64)   -> 16 WMMAs/wave
// Output Y row-major over (b*32+i) x (part*48+k)  (f16)
// ---------------------------------------------------------------------------
__global__ void __launch_bounds__(128, 1)
bc_mix(unsigned int* __restrict__ ws) {
  __shared__ unsigned int sU[2048];
  const int bin = blockIdx.x >> 5;  // 48 bins x 32 column-blocks
  {
    const uint4* src = (const uint4*)(ws + U_OFF + bin * 2048);
    uint4* dstv = (uint4*)sU;
    for (int i = threadIdx.x; i < 512; i += 128) dstv[i] = src[i];
  }
  __syncthreads();

  const int lane = threadIdx.x & 31;
  const int wave = threadIdx.x >> 5;
  const int bt0  = ((blockIdx.x & 31) * 4 + wave) * 2;  // 2 batch-tiles per wave
  const _Float16* X2 = (const _Float16*)(ws + X2_OFF_DW);
  _Float16* Y = (_Float16*)(ws + Y_OFF_DW);

  v8f acc[2][4] = {};
#pragma unroll
  for (int kt = 0; kt < 2; ++kt) {
    v16h a0 = load_A_f16(X2 + bin * 262144 + (bt0 + 0) * 1024 + kt * 32, 64, lane);
    v16h a1 = load_A_f16(X2 + bin * 262144 + (bt0 + 1) * 1024 + kt * 32, 64, lane);
#pragma unroll
    for (int nt = 0; nt < 4; ++nt) {
      v16h b = load_B_frag(sU + (kt * 4 + nt) * 256, lane);
      acc[0][nt] = wmma_f16(a0, b, acc[0][nt]);
      acc[1][nt] = wmma_f16(a1, b, acc[1][nt]);
    }
  }

  const int nc  = lane & 15;
  const int mhi = (lane < 16) ? 0 : 8;
#pragma unroll
  for (int u = 0; u < 2; ++u) {
#pragma unroll
    for (int nt = 0; nt < 4; ++nt) {
      int n = nt * 16 + nc;
      int i = n & 31, part = n >> 5;
#pragma unroll
      for (int v = 0; v < 8; ++v) {
        int bb = (bt0 + u) * 16 + v + mhi;
        Y[(bb * 32 + i) * 96 + part * 48 + bin] = (_Float16)acc[u][nt][v];
      }
    }
  }
}

// ---------------------------------------------------------------------------
// Kernel 3: inverse DFT GEMM. out(131072 x 128) = Y(131072 x 96) @ G(96 x 128)
// G fragments staged in LDS (24 KB); 2 M-tiles per wave -> 48 WMMAs/wave.
// __launch_bounds__(128,1): full VGPR budget for the 128-reg accumulator set.
// Row r = b*32 + i maps to out[b*4096 + i*128 + t] -> contiguous f32 store.
// ---------------------------------------------------------------------------
__global__ void __launch_bounds__(128, 1)
bc_inv_dft(unsigned int* __restrict__ ws, float* __restrict__ out) {
  __shared__ unsigned int sG[6144];
  {
    const uint4* src = (const uint4*)(ws + G_OFF);
    uint4* dstv = (uint4*)sG;
    for (int i = threadIdx.x; i < 1536; i += 128) dstv[i] = src[i];
  }
  __syncthreads();

  const int lane = threadIdx.x & 31;
  const int wave = threadIdx.x >> 5;
  const int mt0  = (blockIdx.x * 4 + wave) * 2;
  const _Float16* Y = (const _Float16*)(ws + Y_OFF_DW);

  __builtin_prefetch(Y + (size_t)mt0 * 1536 + 1536, 0, 1);

  v8f acc[2][8] = {};
#pragma unroll
  for (int kt = 0; kt < 3; ++kt) {
    v16h a0 = load_A_f16(Y + (mt0 + 0) * 1536 + kt * 32, 96, lane);
    v16h a1 = load_A_f16(Y + (mt0 + 1) * 1536 + kt * 32, 96, lane);
#pragma unroll
    for (int nt = 0; nt < 8; ++nt) {
      v16h b = load_B_frag(sG + (kt * 8 + nt) * 256, lane);
      acc[0][nt] = wmma_f16(a0, b, acc[0][nt]);
      acc[1][nt] = wmma_f16(a1, b, acc[1][nt]);
    }
  }

  const int nc  = lane & 15;
  const int mhi = (lane < 16) ? 0 : 8;
#pragma unroll
  for (int u = 0; u < 2; ++u) {
#pragma unroll
    for (int nt = 0; nt < 8; ++nt) {
#pragma unroll
      for (int v = 0; v < 8; ++v) {
        out[((mt0 + u) * 16 + v + mhi) * 128 + nt * 16 + nc] = acc[u][nt][v];
      }
    }
  }
}

extern "C" void kernel_launch(void* const* d_in, const int* in_sizes, int n_in,
                              void* d_out, int out_size, void* d_ws, size_t ws_size,
                              hipStream_t stream) {
  const float* x  = (const float*)d_in[0];   // (4096, 4096) f32
  const float* Wr = (const float*)d_in[1];   // (32, 32, 65) f32
  const float* Wi = (const float*)d_in[2];   // (32, 32, 65) f32
  float* out = (float*)d_out;                // (4096, 4096) f32
  unsigned int* ws = (unsigned int*)d_ws;

  // 0: bake Fb / U_k / G into WMMA B-fragment layout (110592 dwords total)
  bc_build_mats<<<432, 256, 0, stream>>>(Wr, Wi, ws);
  // 1: forward DFT GEMM (8192 M-tiles; 4 waves/block x 2 tiles/wave)
  bc_fwd_dft<<<1024, 128, 0, stream>>>(x, ws);
  // 2: per-bin complex mixing (48 bins x 32 blocks; 4 waves x 2 batch-tiles)
  bc_mix<<<1536, 128, 0, stream>>>(ws);
  // 3: inverse DFT GEMM (8192 M-tiles)
  bc_inv_dft<<<1024, 128, 0, stream>>>(ws, out);
}
